// MaskBBoxPredictor_49959059587442
// MI455X (gfx1250) — matrix operand order repaired and checked by reading, not compiled, not run on hardware
//
#include <hip/hip_runtime.h>

// ---------------------------------------------------------------------------
// MaskBBoxPredictor for MI455X (gfx1250): bf16 WMMA implicit-GEMM convs
// ---------------------------------------------------------------------------

typedef __attribute__((ext_vector_type(16))) __bf16 v16bf;
typedef __attribute__((ext_vector_type(8)))  float  v8f;

union FragU {
    struct { uint4 lo, hi; } q;
    v16bf v;
};

#define TILE_N 256     // spatial positions per workgroup
#define KCH    64      // K elements per mainloop iteration (2 wmma-K)
#define KIDS   20      // object ids 1..20

// ------------------------- layout transform kernels ------------------------

// x: NCHW f32 [B][C][HW] -> NHWC bf16 [B][HW][C], LDS-tiled 32x32 transpose.
// grid (HW/32, C/32, B), block 256 (32x8)
__global__ __launch_bounds__(256) void k_transpose_bf16(
    const float* __restrict__ x, __bf16* __restrict__ out, int C, int HW)
{
    const int b   = blockIdx.z;
    const int sp0 = blockIdx.x * 32;
    const int c0  = blockIdx.y * 32;
    __shared__ float t[32][33];
    const int tx = threadIdx.x & 31;
    const int ty = threadIdx.x >> 5;   // 0..7
    #pragma unroll
    for (int j = 0; j < 4; ++j) {
        int c = c0 + ty + j * 8;
        t[ty + j * 8][tx] = x[((long)b * C + c) * HW + sp0 + tx];
    }
    __syncthreads();
    #pragma unroll
    for (int j = 0; j < 4; ++j) {
        int sp = sp0 + ty + j * 8;
        out[((long)b * HW + sp) * C + c0 + tx] = (__bf16)t[tx][ty + j * 8];
    }
}

// w: OIHW f32 [O][I][3][3] -> bf16 [O][9*I] with k = (r*3+s)*I + i
__global__ __launch_bounds__(256) void k_repack_w(
    const float* __restrict__ w, __bf16* __restrict__ out, int O, int I)
{
    long total = (long)O * I * 9;
    long idx = (long)blockIdx.x * blockDim.x + threadIdx.x;
    long stride = (long)gridDim.x * blockDim.x;
    int K = 9 * I;
    for (; idx < total; idx += stride) {
        int k  = (int)(idx % K);
        int o  = (int)(idx / K);
        int rs = k / I;
        int ci = k - rs * I;
        out[(long)o * K + k] = (__bf16)w[((long)(o * I + ci) * 9) + rs];
    }
}

// ------------------------- WMMA implicit-GEMM conv --------------------------
// in   : NHWC bf16 [B*HW][Cin]
// wrep : bf16 [Cout][Ktot],  Ktot = 9*Cin, k = (r*3+s)*Cin + cin
// out  : NHWC f32 [B*HW][Cout]  (bias fused)
// tile : 16 outC x 256 spatial; 8 waves x (2 N-subtiles); K-chunk 64;
//        ping-pong LDS (one barrier per chunk), register-prefetched globals.
// grid : (B*HW/TILE_N, Cout/16), block 256
__global__ __launch_bounds__(256) void k_conv3x3_wmma_bf16(
    const __bf16* __restrict__ in, const __bf16* __restrict__ wrep,
    const float* __restrict__ bias, float* __restrict__ out,
    int Cin, int Cout, int H, int W)
{
    const int Ktot = 9 * Cin;
    const int HW   = H * W;
    const int nch  = Cin / KCH;          // chunks per tap (Cin multiple of 64)
    const int tid  = threadIdx.x;
    const int lane = tid & 31;
    const int wave = tid >> 5;
    const int n0   = blockIdx.x * TILE_N;
    const int oc0  = blockIdx.y * 16;

    __shared__ __align__(16) __bf16 Atile[2][16 * KCH];      // 2x2 KB  [m][k]
    __shared__ __align__(16) __bf16 Btile[2][TILE_N * KCH];  // 2x32 KB [n][k]

    // ---- loader geometry, hoisted out of the K loop ----
    const int partB = tid & 7;           // 8-bf16 segment within a 64-k row
    const int nlB   = tid >> 3;          // base spatial row (passes add +32)
    int  hB[8], wB[8];
    long baseB[8];
    #pragma unroll
    for (int p = 0; p < 8; ++p) {
        int n  = n0 + nlB + p * 32;
        int b  = n / HW;
        int sp = n - b * HW;
        hB[p] = sp / W;
        wB[p] = sp - hB[p] * W;
        baseB[p] = (long)b * HW * Cin;
    }
    const int  mA    = tid >> 3;         // 0..31 (only <16 active)
    const int  partA = tid & 7;
    const bool doA   = tid < 128;

    uint4 regB[8];
    uint4 regA = {0u, 0u, 0u, 0u};

    auto issue_loads = [&](int dr, int ds, int rsI, int c0) {
        #pragma unroll
        for (int p = 0; p < 8; ++p) {
            const int hh = hB[p] + dr;
            const int ww = wB[p] + ds;
            uint4 v = {0u, 0u, 0u, 0u};
            if (hh >= 0 && hh < H && ww >= 0 && ww < W) {
                v = *(const uint4*)(in + baseB[p] +
                        (long)(hh * W + ww) * Cin + c0 + partB * 8);
            }
            regB[p] = v;
        }
        if (doA) {
            const __bf16* ap = wrep + (long)(oc0 + mA) * Ktot +
                               rsI * Cin + c0 + partA * 8;
            regA = *(const uint4*)ap;
            __builtin_prefetch(ap + Cin, 0, 3);   // warm next tap's A row (L2)
        }
    };
    auto commit = [&](int buf) {
        #pragma unroll
        for (int p = 0; p < 8; ++p)
            ((uint4*)Btile[buf])[(nlB + p * 32) * 8 + partB] = regB[p];
        if (doA)
            ((uint4*)Atile[buf])[mA * 8 + partA] = regA;
    };

    v8f acc0 = {0.f, 0.f, 0.f, 0.f, 0.f, 0.f, 0.f, 0.f};
    v8f acc1 = {0.f, 0.f, 0.f, 0.f, 0.f, 0.f, 0.f, 0.f};

    const int m16 = lane & 15;
    const int hi  = lane >> 4;
    const int rowA = (wave * 32 + m16) * 8;       // subtile-0 row, uint4 units
    const int rowB = (wave * 32 + 16 + m16) * 8;  // subtile-1 row

    // ---- prologue: chunk 0 into buffer 0 ----
    issue_loads(-1, -1, 0, 0);
    commit(0);
    __syncthreads();

    int dr = -1, ds = -1, rsI = 0, c0 = 0;
    const int iters = 9 * nch;
    for (int it = 0; it < iters; ++it) {
        const int cur = it & 1;
        // ---- advance tap/chunk state without division ----
        int c1 = c0 + KCH, dr1 = dr, ds1 = ds, rsI1 = rsI;
        if (c1 == Cin) {
            c1 = 0; rsI1 = rsI + 1;
            ds1 = ds + 1;
            if (ds1 == 2) { ds1 = -1; dr1 = dr + 1; }
        }
        const bool more = (it + 1 < iters);
        if (more) issue_loads(dr1, ds1, rsI1, c1);   // overlap with WMMAs below
        dr = dr1; ds = ds1; rsI = rsI1; c0 = c1;

        // ---- fragments in contiguous k order (k = 16*hi + s, A and B alike), 4 WMMAs ----
        const uint4* Au = (const uint4*)Atile[cur];
        const uint4* Bu = (const uint4*)Btile[cur];
        FragU a0, a1, b00, b01, b10, b11;
        a0.q.lo  = Au[m16 * 8 + hi * 2];
        a0.q.hi  = Au[m16 * 8 + hi * 2 + 1];
        a1.q.lo  = Au[m16 * 8 + 4 + hi * 2];
        a1.q.hi  = Au[m16 * 8 + 4 + hi * 2 + 1];
        b00.q.lo = Bu[rowA + hi * 2];
        b00.q.hi = Bu[rowA + hi * 2 + 1];
        b01.q.lo = Bu[rowA + 4 + hi * 2];
        b01.q.hi = Bu[rowA + 4 + hi * 2 + 1];
        b10.q.lo = Bu[rowB + hi * 2];
        b10.q.hi = Bu[rowB + hi * 2 + 1];
        b11.q.lo = Bu[rowB + 4 + hi * 2];
        b11.q.hi = Bu[rowB + 4 + hi * 2 + 1];

        acc0 = __builtin_amdgcn_wmma_f32_16x16x32_bf16(
            false, a0.v, false, b00.v, (short)0, acc0, false, false);
        acc0 = __builtin_amdgcn_wmma_f32_16x16x32_bf16(
            false, a1.v, false, b01.v, (short)0, acc0, false, false);
        acc1 = __builtin_amdgcn_wmma_f32_16x16x32_bf16(
            false, a0.v, false, b10.v, (short)0, acc1, false, false);
        acc1 = __builtin_amdgcn_wmma_f32_16x16x32_bf16(
            false, a1.v, false, b11.v, (short)0, acc1, false, false);

        // ---- commit chunk it+1 into the other buffer; single barrier ----
        if (more) commit(cur ^ 1);
        __syncthreads();
    }

    // ---- epilogue: D layout lane=column n, VGPR i -> m = i + 8*hi ----
    const int   m0 = hi * 8;
    const long  nOut0 = (long)n0 + wave * 32 + m16;
    const long  nOut1 = nOut0 + 16;
    const float4 bi0 = *(const float4*)(bias + oc0 + m0);
    const float4 bi1 = *(const float4*)(bias + oc0 + m0 + 4);
    float4 o0, o1;
    o0.x = acc0[0] + bi0.x; o0.y = acc0[1] + bi0.y;
    o0.z = acc0[2] + bi0.z; o0.w = acc0[3] + bi0.w;
    o1.x = acc0[4] + bi1.x; o1.y = acc0[5] + bi1.y;
    o1.z = acc0[6] + bi1.z; o1.w = acc0[7] + bi1.w;
    float* dst0 = out + nOut0 * Cout + oc0 + m0;
    *(float4*)dst0       = o0;
    *(float4*)(dst0 + 4) = o1;
    o0.x = acc1[0] + bi0.x; o0.y = acc1[1] + bi0.y;
    o0.z = acc1[2] + bi0.z; o0.w = acc1[3] + bi0.w;
    o1.x = acc1[4] + bi1.x; o1.y = acc1[5] + bi1.y;
    o1.z = acc1[6] + bi1.z; o1.w = acc1[7] + bi1.w;
    float* dst1 = out + nOut1 * Cout + oc0 + m0;
    *(float4*)dst1       = o0;
    *(float4*)(dst1 + 4) = o1;
}

// ------------------------- GroupNorm + ReLU ---------------------------------

// one block per (b,g): reduce sum/sumsq over HW * (C/G) elements of NHWC f32.
// 256 % Cg == 0 holds for all layers (Cg = 32 or 16).
__global__ __launch_bounds__(256) void k_gn_stats(
    const float* __restrict__ conv, float* __restrict__ stats,
    int C, int G, int HW)
{
    const int bg = blockIdx.x;
    const int b  = bg / G;
    const int g  = bg - b * G;
    const int Cg = C / G;
    const int c0 = g * Cg;
    const int tid = threadIdx.x;

    const int c      = tid % Cg;
    const int sp0    = tid / Cg;
    const int spstep = 256 / Cg;

    float s = 0.f, ss = 0.f;
    for (int sp = sp0; sp < HW; sp += spstep) {
        float v = conv[((long)b * HW + sp) * C + c0 + c];
        s += v; ss += v * v;
    }
    __shared__ float rs_[256], rss[256];
    rs_[tid] = s; rss[tid] = ss;
    __syncthreads();
    for (int off = 128; off > 0; off >>= 1) {
        if (tid < off) {
            rs_[tid] += rs_[tid + off];
            rss[tid] += rss[tid + off];
        }
        __syncthreads();
    }
    if (tid == 0) {
        float total = (float)HW * (float)Cg;
        float inv   = 1.f / total;
        float mean  = rs_[0] * inv;
        float var   = rss[0] * inv - mean * mean;  // biased, matches jnp.var
        stats[bg * 2 + 0] = mean;
        stats[bg * 2 + 1] = rsqrtf(var + 1e-5f);
    }
}

// normalize + affine + relu; optionally write bf16 (next conv) and/or f32 (pooling)
__global__ __launch_bounds__(256) void k_gn_norm_relu(
    const float* __restrict__ conv, const float* __restrict__ stats,
    const float* __restrict__ gw, const float* __restrict__ gb,
    __bf16* __restrict__ out_bf, float* __restrict__ out_f,
    int C, int G, int HW, long total /* B*HW*C */)
{
    const int Cg = C / G;
    long i = (long)blockIdx.x * blockDim.x + threadIdx.x;
    long stride = (long)gridDim.x * blockDim.x;
    for (; i < total; i += stride) {
        int  c  = (int)(i % C);
        long sp = i / C;
        int  b  = (int)(sp / HW);
        int  g  = c / Cg;
        float mean = stats[(b * G + g) * 2 + 0];
        float rstd = stats[(b * G + g) * 2 + 1];
        float v = (conv[i] - mean) * rstd * gw[c] + gb[c];
        v = fmaxf(v, 0.f);
        if (out_bf) out_bf[i] = (__bf16)v;
        if (out_f)  out_f[i]  = v;
    }
}

// ------------------------- segment pooling + heads --------------------------

__global__ __launch_bounds__(256) void k_zero(float* __restrict__ p, int n)
{
    int i = blockIdx.x * blockDim.x + threadIdx.x;
    if (i < n) p[i] = 0.f;
}

// h: NHWC f32 [B*HW][64]; masks: [B][HW] int32.
// grid (chunks, B); LDS-accumulate then global atomicAdd.
__global__ __launch_bounds__(256) void k_pool_seg(
    const float* __restrict__ h, const int* __restrict__ masks,
    float* __restrict__ sums, float* __restrict__ counts,
    int HW, int chunk)
{
    const int b  = blockIdx.y;
    const int p0 = blockIdx.x * chunk;
    __shared__ float ls[KIDS * 64];
    __shared__ float lc[KIDS];
    for (int i = threadIdx.x; i < KIDS * 64; i += 256) ls[i] = 0.f;
    if (threadIdx.x < KIDS) lc[threadIdx.x] = 0.f;
    __syncthreads();

    const int c   = threadIdx.x & 63;
    const int sub = threadIdx.x >> 6;  // 4 pixels per iteration
    for (int p = p0 + sub; p < p0 + chunk; p += 4) {
        int id = masks[(long)b * HW + p];
        if (id > 0) {
            int k = id - 1;
            atomicAdd(&ls[k * 64 + c], h[((long)b * HW + p) * 64 + c]);
            if (c == 0) atomicAdd(&lc[k], 1.f);
        }
    }
    __syncthreads();
    for (int i = threadIdx.x; i < KIDS * 64; i += 256)
        atomicAdd(&sums[(long)b * KIDS * 64 + i], ls[i]);
    if (threadIdx.x < KIDS)
        atomicAdd(&counts[b * KIDS + threadIdx.x], lc[threadIdx.x]);
}

// pooled = sums/(counts+1e-6);  boxes = pooled@wb^T+bb;  scores = sigmoid(...)
// out: [560 boxes][80 scores] f32 flat
__global__ __launch_bounds__(128) void k_heads(
    const float* __restrict__ sums, const float* __restrict__ counts,
    const float* __restrict__ wb, const float* __restrict__ bb,
    const float* __restrict__ wc, const float* __restrict__ bc,
    float* __restrict__ out)
{
    int t = threadIdx.x;
    if (t >= 4 * KIDS) return;
    float inv = 1.f / (counts[t] + 1e-6f);
    float pooled[64];
    #pragma unroll
    for (int c = 0; c < 64; ++c) pooled[c] = sums[t * 64 + c] * inv;
    #pragma unroll
    for (int o = 0; o < 7; ++o) {
        float a = bb[o];
        for (int c = 0; c < 64; ++c) a += pooled[c] * wb[o * 64 + c];
        out[t * 7 + o] = a;
    }
    float s = bc[0];
    for (int c = 0; c < 64; ++c) s += pooled[c] * wc[c];
    out[4 * KIDS * 7 + t] = 1.f / (1.f + expf(-s));
}

// ---------------------------------------------------------------------------

extern "C" void kernel_launch(void* const* d_in, const int* in_sizes, int n_in,
                              void* d_out, int out_size, void* d_ws, size_t ws_size,
                              hipStream_t stream) {
    (void)in_sizes; (void)n_in; (void)out_size; (void)ws_size;
    const int Bn = 4, H = 128, W = 128, HW = H * W;
    const long totSp = (long)Bn * HW;  // 65536

    const float* x     = (const float*)d_in[0];
    const int*   masks = (const int*)  d_in[1];
    const float* w1 = (const float*)d_in[2];  const float* b1 = (const float*)d_in[3];
    const float* g1w = (const float*)d_in[4]; const float* g1b = (const float*)d_in[5];
    const float* w2 = (const float*)d_in[6];  const float* b2 = (const float*)d_in[7];
    const float* g2w = (const float*)d_in[8]; const float* g2b = (const float*)d_in[9];
    const float* w3 = (const float*)d_in[10]; const float* b3 = (const float*)d_in[11];
    const float* g3w = (const float*)d_in[12];const float* g3b = (const float*)d_in[13];
    const float* wb = (const float*)d_in[14]; const float* bb = (const float*)d_in[15];
    const float* wc = (const float*)d_in[16]; const float* bc = (const float*)d_in[17];
    float* outp = (float*)d_out;

    // workspace layout (bytes)
    char* ws = (char*)d_ws;
    size_t o = 0;
    __bf16* bufA_bf = (__bf16*)(ws + o); o += (size_t)totSp * 512 * 2;      // 67 MB
    __bf16* bufB_bf = (__bf16*)(ws + o);
    float*  bufB_f  = (float*) bufB_bf;  o += (size_t)totSp * 256 * 2;      // 33.5 MB
    float*  convf   = (float*) (ws + o); o += (size_t)totSp * 256 * 4;      // 67 MB
    __bf16* w1p = (__bf16*)(ws + o); o += (size_t)256 * 4608 * 2;
    __bf16* w2p = (__bf16*)(ws + o); o += (size_t)128 * 2304 * 2;
    __bf16* w3p = (__bf16*)(ws + o); o += (size_t)64  * 1152 * 2;
    float* stats  = (float*)(ws + o); o += 1024;
    float* sums   = (float*)(ws + o); o += (size_t)Bn * KIDS * 64 * 4;
    float* counts = (float*)(ws + o); o += (size_t)Bn * KIDS * 4;

    // --- layout transforms ---
    k_transpose_bf16<<<dim3(HW / 32, 512 / 32, Bn), 256, 0, stream>>>(x, bufA_bf, 512, HW);
    k_repack_w<<<1024, 256, 0, stream>>>(w1, w1p, 256, 512);
    k_repack_w<<<512,  256, 0, stream>>>(w2, w2p, 128, 256);
    k_repack_w<<<256,  256, 0, stream>>>(w3, w3p, 64, 128);
    k_zero<<<(Bn * KIDS * 65 + 255) / 256, 256, 0, stream>>>(sums, Bn * KIDS * 64);
    k_zero<<<1, 256, 0, stream>>>(counts, Bn * KIDS);

    dim3 blk(256);
    // --- layer 1: 512 -> 256, GN(8) ---
    k_conv3x3_wmma_bf16<<<dim3(totSp / TILE_N, 256 / 16), blk, 0, stream>>>(
        bufA_bf, w1p, b1, convf, 512, 256, H, W);
    k_gn_stats<<<Bn * 8, blk, 0, stream>>>(convf, stats, 256, 8, HW);
    k_gn_norm_relu<<<2048, blk, 0, stream>>>(convf, stats, g1w, g1b,
        bufB_bf, (float*)nullptr, 256, 8, HW, totSp * 256);

    // --- layer 2: 256 -> 128, GN(4) ---
    k_conv3x3_wmma_bf16<<<dim3(totSp / TILE_N, 128 / 16), blk, 0, stream>>>(
        bufB_bf, w2p, b2, convf, 256, 128, H, W);
    k_gn_stats<<<Bn * 4, blk, 0, stream>>>(convf, stats, 128, 4, HW);
    k_gn_norm_relu<<<2048, blk, 0, stream>>>(convf, stats, g2w, g2b,
        bufA_bf, (float*)nullptr, 128, 4, HW, totSp * 128);

    // --- layer 3: 128 -> 64, GN(4) ---
    k_conv3x3_wmma_bf16<<<dim3(totSp / TILE_N, 64 / 16), blk, 0, stream>>>(
        bufA_bf, w3p, b3, convf, 128, 64, H, W);
    k_gn_stats<<<Bn * 4, blk, 0, stream>>>(convf, stats, 64, 4, HW);
    k_gn_norm_relu<<<2048, blk, 0, stream>>>(convf, stats, g3w, g3b,
        (__bf16*)nullptr, bufB_f, 64, 4, HW, totSp * 64);

    // --- segment pooling + heads ---
    k_pool_seg<<<dim3(32, Bn), blk, 0, stream>>>(bufB_f, masks, sums, counts, HW, HW / 32);
    k_heads<<<1, 128, 0, stream>>>(sums, counts, wb, bb, wc, bc, outp);
}
